// TextEncoder_16587163697791
// MI455X (gfx1250) — matrix operand order repaired and checked by reading, not compiled, not run on hardware
//
#include <hip/hip_runtime.h>
#include <hip/hip_bf16.h>

// MI455X / gfx1250 implementation.
//  - bf16 WMMA (v_wmma_f32_16x16x32_bf16, fp32 accum) for both GRU GEMMs
//  - persistent per-16-row recurrence blocks; fp32 hidden master + bf16
//    LDS mirror so A-fragments are two ds_load_b128 (no cvt on critical path)
//  - fused ragged packing (out_rfts / out_embs) and mask generation
//  - small GCN done with plain VALU (2 GFLOP, irrelevant to roofline)

typedef __attribute__((ext_vector_type(16))) __bf16 v16bf;
typedef __attribute__((ext_vector_type(8)))  float  v8f;

namespace {
constexpr int kB3   = 192;    // BS*NSEG
constexpr int kSLEN = 256;
constexpr int kE    = 300;
constexpr int kEP   = 320;    // E padded to multiple of 32 (WMMA K)
constexpr int kH    = 512;
constexpr int kG    = 1536;   // 3H
constexpr int kBS   = 64;
constexpr int kTT   = 768;    // NSEG*SLEN packed length
constexpr int kNN   = 18;     // GCN nodes
constexpr int kNF   = 600;
constexpr int kNHID = 600;
constexpr int kNOUT = 1024;

// workspace byte offsets (16B aligned)
constexpr size_t WIH_OFF = 0;                                         // bf16 [1536][320]
constexpr size_t WHH_OFF = WIH_OFF + (size_t)kG * kEP * 2;            // bf16 [1536][512]
constexpr size_t TOK_OFF = WHH_OFF + (size_t)kG * kH * 2;             // bf16 [49152][320]
constexpr size_t XI_OFF  = TOK_OFF + (size_t)kB3 * kSLEN * kEP * 2;   // f32  [49152][1536]
constexpr size_t T1_OFF  = XI_OFF  + (size_t)kB3 * kSLEN * kG * 4;    // f32  [1152][600]
constexpr size_t H1_OFF  = T1_OFF  + (size_t)kBS * kNN * kNHID * 4;   // f32  [1152][600]
constexpr size_t T2_OFF  = H1_OFF  + (size_t)kBS * kNN * kNHID * 4;   // f32  [1152][1024]

// d_out float offsets (concatenated return order)
constexpr size_t O_RFTS = 0;
constexpr size_t O_EMBS = O_RFTS + (size_t)kBS * kTT * kH;
constexpr size_t O_MSKS = O_EMBS + (size_t)kBS * kTT * kE;
constexpr size_t O_SG   = O_MSKS + (size_t)kBS * kTT;
constexpr size_t O_HIDS = O_SG   + (size_t)kBS * kNN * kNOUT;
constexpr size_t O_TOT  = O_HIDS + (size_t)kBS * 3 * kH;
} // namespace

// ---------------------------------------------------------------------------
// WMMA fragment helpers (wave32 lane layouts per CDNA5 ISA 7.12.2)
// ---------------------------------------------------------------------------

// A (16x32 bf16, row-major, lda elements): lane l holds row (l&15),
// K = (l>>4)*8 + {0..7} in v0..3 and 16 + (l>>4)*8 + {0..7} in v4..7.
__device__ inline v16bf load_a_g(const __bf16* __restrict__ A, int lda) {
  const int l = threadIdx.x & 31;
  const __bf16* p = A + (size_t)(l & 15) * lda + ((l >> 4) << 3);
  union { v16bf v; uint4 q[2]; } u;
  u.q[0] = *(const uint4*)p;         // K+0..7   (16B)
  u.q[1] = *(const uint4*)(p + 16);  // K+16..23 (16B)
  return u.v;
}

// Same fragment shape sourced from LDS (two ds_load_b128).
__device__ inline v16bf load_a_lds(const __bf16* hbf, int ktile) {
  const int l = threadIdx.x & 31;
  const __bf16* p = hbf + (l & 15) * kH + ktile * 32 + ((l >> 4) << 3);
  union { v16bf v; uint4 q[2]; } u;
  u.q[0] = *(const uint4*)p;
  u.q[1] = *(const uint4*)(p + 16);
  return u.v;
}

// B (32x16 bf16): lane l holds column N=(l&15), K = (l>>4)*8 + {0..7} and 16 + (l>>4)*8 + {0..7}.
// Column-major B == row-major B^T == the torch weight [out][in] itself.
__device__ inline v16bf load_b_g(const __bf16* __restrict__ Bt, int ldb) {
  const int l = threadIdx.x & 31;
  const __bf16* p = Bt + (size_t)(l & 15) * ldb + ((l >> 4) << 3);
  union { v16bf v; uint4 q[2]; } u;
  u.q[0] = *(const uint4*)p;         // 16B
  u.q[1] = *(const uint4*)(p + 16);  // 16B
  return u.v;
}

// D/C (16x16 f32): VGPR r, lane l -> M = r + (l>>4)*8, N = (l&15).
__device__ inline void store_tile(float* __restrict__ C, int ldc, v8f d) {
  const int l  = threadIdx.x & 31;
  const int cb = l & 15;
  const int rb = (l >> 4) * 8;
#pragma unroll
  for (int r = 0; r < 8; ++r) C[(size_t)(rb + r) * ldc + cb] = d[r];
}

__device__ inline v8f wmma_bf16(v16bf a, v16bf b, v8f c) {
  return __builtin_amdgcn_wmma_f32_16x16x32_bf16(
      /*neg_a=*/false, a, /*neg_b=*/false, b,
      /*c_mod=*/(short)0, c, /*reuse_a=*/false, /*reuse_b=*/false);
}

// ---------------------------------------------------------------------------
// Kernels
// ---------------------------------------------------------------------------

__global__ void zero_kernel(float* __restrict__ out, size_t n) {
  for (size_t i = (size_t)blockIdx.x * blockDim.x + threadIdx.x; i < n;
       i += (size_t)gridDim.x * blockDim.x)
    out[i] = 0.0f;
}

// fp32 weights -> bf16 (w_ih K-padded 300->320 with zeros)
__global__ void prep_w_kernel(const float* __restrict__ wih,
                              const float* __restrict__ whh,
                              __bf16* __restrict__ wihb,
                              __bf16* __restrict__ whhb) {
  const int n1 = kG * kEP;
  const int n2 = kG * kH;
  for (int i = blockIdx.x * blockDim.x + threadIdx.x; i < n1 + n2;
       i += gridDim.x * blockDim.x) {
    if (i < n1) {
      const int g = i / kEP, k = i % kEP;
      wihb[i] = (__bf16)((k < kE) ? wih[(size_t)g * kE + k] : 0.0f);
    } else {
      const int j = i - n1;
      whhb[j] = (__bf16)whh[j];
    }
  }
}

// Embedding gather -> bf16 tokens (padded K) + packed out_embs scatter.
__global__ void embed_kernel(const int* __restrict__ inds,
                             const int* __restrict__ lens,
                             const float* __restrict__ emb,
                             __bf16* __restrict__ tokb,
                             float* __restrict__ out_embs) {
  const int bt  = blockIdx.x;          // b3*SLEN + t
  const int b3  = bt >> 8;
  const int t   = bt & (kSLEN - 1);
  const int bb  = b3 / 3, seg = b3 % 3;
  const int l0 = lens[bb * 3 + 0], l1 = lens[bb * 3 + 1], l2 = lens[bb * 3 + 2];
  const int mylen = (seg == 0) ? l0 : (seg == 1 ? l1 : l2);
  const int myoff = (seg == 0) ? 0 : (seg == 1 ? l0 : l0 + l1);
  const bool valid = t < mylen;

  const float* erow = emb + (size_t)inds[bt] * kE;
  __bf16* trow = tokb + (size_t)bt * kEP;
  float* orow = out_embs + ((size_t)bb * kTT + myoff + t) * kE;

  for (int e = threadIdx.x; e < kEP; e += blockDim.x) {
    const float v = (e < kE) ? erow[e] : 0.0f;
    trow[e] = (__bf16)v;
    if (valid && e < kE) orow[e] = v;
  }
}

// xi = tok @ w_ih^T + b_ih : [49152,320]bf16 x [320,1536] -> f32.
// 8 waves: 4 M-tiles x 2 N-groups of 4 tiles => 64x128 block tile.
__global__ void __launch_bounds__(256) gemm_xi_kernel(
    const __bf16* __restrict__ tokb, const __bf16* __restrict__ wihb,
    const float* __restrict__ b_ih, float* __restrict__ xi) {
  const int wid    = threadIdx.x >> 5;
  const int mrow   = blockIdx.x * 64 + (wid & 3) * 16;
  const int ntile0 = blockIdx.y * 8 + (wid >> 2) * 4;

  v8f acc[4];
#pragma unroll
  for (int j = 0; j < 4; ++j) acc[j] = v8f{};

  const __bf16* Abase = tokb + (size_t)mrow * kEP;
#pragma unroll
  for (int kt = 0; kt < kEP / 32; ++kt) {
    if (kt + 1 < kEP / 32)
      __builtin_prefetch(Abase + (kt + 1) * 32, 0, 3);  // global_prefetch_b8
    const v16bf a = load_a_g(Abase + kt * 32, kEP);
#pragma unroll
    for (int j = 0; j < 4; ++j) {
      const v16bf b =
          load_b_g(wihb + (size_t)(ntile0 + j) * 16 * kEP + kt * 32, kEP);
      acc[j] = wmma_bf16(a, b, acc[j]);
    }
  }

  const int l  = threadIdx.x & 31;
  const int cb = l & 15;
  const int rb = (l >> 4) * 8;
#pragma unroll
  for (int j = 0; j < 4; ++j) {
    const int nt = ntile0 + j;
    const float bias = b_ih[nt * 16 + cb];
    float* C = xi + (size_t)mrow * kG + nt * 16;
#pragma unroll
    for (int r = 0; r < 8; ++r)
      C[(size_t)(rb + r) * kG + cb] = acc[j][r] + bias;
  }
}

// Persistent GRU recurrence: block owns 16 batch rows, h lives in LDS
// (fp32 master for the carry + bf16 mirror feeding WMMA A-fragments).
// Per step: gh = h @ w_hh^T via WMMA (96 N-tiles / 16 waves), then gates,
// length-masked update + packed scatter of ys, final h -> out_hids.
__global__ void __launch_bounds__(512) gru_kernel(
    const float* __restrict__ xi, const __bf16* __restrict__ whhb,
    const float* __restrict__ b_hh, const int* __restrict__ lens,
    float* __restrict__ out_rfts, float* __restrict__ out_hids) {
  __shared__ __align__(16) float  hbuf[16 * kH];  // 32 KB fp32 hidden (carry)
  __shared__ __align__(16) __bf16 hbf[16 * kH];   // 16 KB bf16 mirror (WMMA A)
  __shared__ __align__(16) float  gh[16 * kG];    // 96 KB recurrent pre-acts

  const int tid = threadIdx.x;
  const int wid = tid >> 5;
  const int m0  = blockIdx.x * 16;  // first b3 row of this block

  for (int i = tid; i < 16 * kH; i += blockDim.x) {
    hbuf[i] = 0.0f;
    hbf[i]  = (__bf16)0.0f;
  }

  // gate-phase per-thread constants: row = tid>>5, 16 cols strided by 32
  const int grow  = tid >> 5;       // 0..15
  const int gcol0 = tid & 31;
  const int b3    = m0 + grow;
  const int bb    = b3 / 3, seg = b3 % 3;
  const int l0 = lens[bb * 3 + 0], l1 = lens[bb * 3 + 1], l2 = lens[bb * 3 + 2];
  const int mylen = (seg == 0) ? l0 : (seg == 1 ? l1 : l2);
  const int myoff = (seg == 0) ? 0 : (seg == 1 ? l0 : l0 + l1);
  __syncthreads();

  for (int t = 0; t < kSLEN; ++t) {
    // ---- gh = h @ w_hh^T  (wave handles 6 of 96 N-tiles) ----
    v8f acc[6];
#pragma unroll
    for (int j = 0; j < 6; ++j) acc[j] = v8f{};
#pragma unroll 4
    for (int kt = 0; kt < kH / 32; ++kt) {
      const v16bf a = load_a_lds(hbf, kt);   // two ds_load_b128, no cvt
#pragma unroll
      for (int j = 0; j < 6; ++j) {
        const int nt = wid * 6 + j;
        const v16bf b =
            load_b_g(whhb + (size_t)nt * 16 * kH + kt * 32, kH);
        acc[j] = wmma_bf16(a, b, acc[j]);
      }
    }
#pragma unroll
    for (int j = 0; j < 6; ++j) store_tile(gh + (wid * 6 + j) * 16, kG, acc[j]);
    __syncthreads();

    // ---- gates + masked update + packed scatter ----
    const float* xip = xi + ((size_t)b3 * kSLEN + t) * kG;
    const bool valid = t < mylen;
    float* orow = out_rfts + ((size_t)bb * kTT + myoff + t) * kH;
#pragma unroll 4
    for (int i = 0; i < 16; ++i) {
      const int c = gcol0 + (i << 5);
      const float hr = gh[grow * kG + c]            + b_hh[c];
      const float hz = gh[grow * kG + kH + c]       + b_hh[kH + c];
      const float hn = gh[grow * kG + 2 * kH + c]   + b_hh[2 * kH + c];
      const float xr = xip[c], xz = xip[kH + c], xn = xip[2 * kH + c];
      const float rg = 1.0f / (1.0f + __expf(-(xr + hr)));
      const float zg = 1.0f / (1.0f + __expf(-(xz + hz)));
      const float ng = tanhf(xn + rg * hn);
      const float hold = hbuf[grow * kH + c];
      const float hnew = (1.0f - zg) * ng + zg * hold;
      const float hout = valid ? hnew : hold;
      hbuf[grow * kH + c] = hout;
      hbf[grow * kH + c]  = (__bf16)hout;
      if (valid) orow[c] = hout;
    }
    __syncthreads();  // protect hbuf/hbf for next step's A-fragment loads
  }

  // final (frozen) hidden per segment
  float* hid = out_hids + (size_t)(bb * 3 + seg) * kH;
#pragma unroll 4
  for (int i = 0; i < 16; ++i) {
    const int c = gcol0 + (i << 5);
    hid[c] = hbuf[grow * kH + c];
  }
}

__global__ void masks_kernel(const int* __restrict__ lens,
                             float* __restrict__ out_msks) {
  const int b = blockIdx.x;
  const int tlen = lens[b * 3] + lens[b * 3 + 1] + lens[b * 3 + 2];
  for (int t = threadIdx.x; t < kTT; t += blockDim.x)
    out_msks[(size_t)b * kTT + t] = (t < tlen) ? 1.0f : 0.0f;
}

// GCN: Y[row] = X[row] @ W  (X row staged in LDS; ~2 GFLOP total -> VALU)
__global__ void gcn_xw_kernel(const float* __restrict__ X,
                              const float* __restrict__ W,
                              float* __restrict__ Y, int K, int N) {
  __shared__ float xs[kNF];
  const int row = blockIdx.x;
  const float* xr = X + (size_t)row * K;
  for (int k = threadIdx.x; k < K; k += blockDim.x) xs[k] = xr[k];
  __syncthreads();
  for (int j = threadIdx.x; j < N; j += blockDim.x) {
    float a = 0.0f;
    for (int k = 0; k < K; ++k) a = fmaf(xs[k], W[(size_t)k * N + j], a);
    Y[(size_t)row * N + j] = a;
  }
}

// GCN: Y[b] = (relu?)(adj[b] @ T[b] + bias) with adj,T staged in LDS
__global__ void gcn_adj_kernel(const float* __restrict__ adj,
                               const float* __restrict__ Tin,
                               const float* __restrict__ bias,
                               float* __restrict__ Y, int N, int do_relu) {
  __shared__ float tb[kNN * kNOUT];  // up to 72 KB
  __shared__ float am[kNN * kNN];
  const int b = blockIdx.x;
  const float* A  = adj + (size_t)b * kNN * kNN;
  const float* Tb = Tin + (size_t)b * kNN * N;
  for (int i = threadIdx.x; i < kNN * kNN; i += blockDim.x) am[i] = A[i];
  for (int i = threadIdx.x; i < kNN * N; i += blockDim.x) tb[i] = Tb[i];
  __syncthreads();
  for (int i = threadIdx.x; i < kNN * N; i += blockDim.x) {
    const int n = i / N, j = i % N;
    float a = bias[j];
#pragma unroll
    for (int m = 0; m < kNN; ++m) a = fmaf(am[n * kNN + m], tb[m * N + j], a);
    if (do_relu) a = fmaxf(a, 0.0f);
    Y[(size_t)b * kNN * N + i] = a;
  }
}

// ---------------------------------------------------------------------------

extern "C" void kernel_launch(void* const* d_in, const int* in_sizes, int n_in,
                              void* d_out, int out_size, void* d_ws,
                              size_t ws_size, hipStream_t stream) {
  const int*   inds = (const int*)d_in[0];
  const int*   lens = (const int*)d_in[1];
  const float* emb  = (const float*)d_in[2];
  const float* w_ih = (const float*)d_in[3];
  const float* w_hh = (const float*)d_in[4];
  const float* b_ih = (const float*)d_in[5];
  const float* b_hh = (const float*)d_in[6];
  const float* sg_x = (const float*)d_in[7];
  const float* adj  = (const float*)d_in[8];
  const float* gw1  = (const float*)d_in[9];
  const float* gb1  = (const float*)d_in[10];
  const float* gw2  = (const float*)d_in[11];
  const float* gb2  = (const float*)d_in[12];
  float* out = (float*)d_out;

  char* ws = (char*)d_ws;
  __bf16* wihb = (__bf16*)(ws + WIH_OFF);
  __bf16* whhb = (__bf16*)(ws + WHH_OFF);
  __bf16* tokb = (__bf16*)(ws + TOK_OFF);
  float*  xi   = (float*)(ws + XI_OFF);
  float*  t1   = (float*)(ws + T1_OFF);
  float*  h1w  = (float*)(ws + H1_OFF);
  float*  t2   = (float*)(ws + T2_OFF);

  zero_kernel<<<4096, 256, 0, stream>>>(out, O_TOT);
  prep_w_kernel<<<1024, 256, 0, stream>>>(w_ih, w_hh, wihb, whhb);
  embed_kernel<<<kB3 * kSLEN, 128, 0, stream>>>(inds, lens, emb, tokb,
                                                out + O_EMBS);
  dim3 gx(kB3 * kSLEN / 64, kG / 128);
  gemm_xi_kernel<<<gx, 256, 0, stream>>>(tokb, wihb, b_ih, xi);
  gru_kernel<<<kB3 / 16, 512, 0, stream>>>(xi, whhb, b_hh, lens,
                                           out + O_RFTS, out + O_HIDS);
  masks_kernel<<<kBS, 256, 0, stream>>>(lens, out + O_MSKS);

  gcn_xw_kernel<<<kBS * kNN, 256, 0, stream>>>(sg_x, gw1, t1, kNF, kNHID);
  gcn_adj_kernel<<<kBS, 256, 0, stream>>>(adj, t1, gb1, h1w, kNHID, 1);
  gcn_xw_kernel<<<kBS * kNN, 256, 0, stream>>>(h1w, gw2, t2, kNHID, kNOUT);
  gcn_adj_kernel<<<kBS, 256, 0, stream>>>(adj, t2, gb2, out + O_SG, kNOUT, 0);
}
